// ProductVectorQuantize_12137577578697
// MI455X (gfx1250) — matrix operand order repaired and checked by reading, not compiled, not run on hardware
//
#include <hip/hip_runtime.h>
#include <math.h>

// ---------------- problem constants ----------------
constexpr int B_ = 32, H_ = 8, W_ = 2000, C_ = 64;
constexpr int OV_ = 4, G_ = 8, CD_ = 8, KC_ = 1024;
constexpr int FIX_ = H_ * C_;          // 512
constexpr int D_ = FIX_ * OV_ / G_;    // 256
constexpr int T_ = W_ / OV_;           // 500
constexpr int HW_ = H_ * W_;           // 16000
constexpr int R_ = B_ * T_;            // 16000 rows per group

constexpr int NWAVE = 4;                       // waves per block
constexpr int ROWS_PER_BLK = NWAVE * 16;       // 64
constexpr int BLKS_PER_G = R_ / ROWS_PER_BLK;  // 250
constexpr int NBLK = G_ * BLKS_PER_G;          // 2000
constexpr int NPART = NBLK * NWAVE;            // 8000 loss partials

// output layout (flat f32, reference return order)
constexpr size_t OUT_ZQ  = 0;
constexpr size_t OUT_ZED = (size_t)B_ * HW_ * C_;                 // 32768000
constexpr size_t OUT_IDX = OUT_ZED + (size_t)B_ * G_ * T_ * CD_;  // +1024000
constexpr size_t OUT_CL  = OUT_IDX + (size_t)B_ * G_ * T_;        // +128000
constexpr size_t OUT_CB  = OUT_CL + 1;

// workspace layout (bytes)
constexpr size_t WS_CBH  = 0;                                      // f16 normalized codebook
constexpr size_t WS_CBF  = WS_CBH + (size_t)G_ * KC_ * CD_ * 2;    // f32 normalized codebook
constexpr size_t WS_CC   = WS_CBF + (size_t)G_ * KC_ * CD_ * 4;    // |c|^2 per code
constexpr size_t WS_WDN  = WS_CC  + (size_t)G_ * KC_ * 4;          // f16 W_down^T  [g][cd][k]
constexpr size_t WS_WUP  = WS_WDN + (size_t)G_ * CD_ * D_ * 2;     // f16 W_up^T    [g][d][k]
constexpr size_t WS_LOSS = WS_WUP + (size_t)G_ * D_ * CD_ * 2;     // per-wave partials

typedef _Float16 h16v __attribute__((ext_vector_type(16)));
typedef _Float16 h8v  __attribute__((ext_vector_type(8)));
typedef _Float16 h2v  __attribute__((ext_vector_type(2)));
typedef float    f8v  __attribute__((ext_vector_type(8)));

// per-wave LDS region: 8192B in/out union + 512B z_down + 64B codes
constexpr int WAVE_BYTES = 8192 + 512 + 64;   // 8768
constexpr int SMEM_BYTES = 16384 + 4096 + NWAVE * WAVE_BYTES;  // 55552

#define SHUF16(lo, hi) __builtin_shufflevector(lo, hi, 0,1,2,3,4,5,6,7,8,9,10,11,12,13,14,15)

// ---------------- kernel 0: prep (normalize codebook, transpose weights to f16) ----
__global__ void pvq_prep(const float* __restrict__ Wd, const float* __restrict__ Wu,
                         const float* __restrict__ cb,
                         _Float16* __restrict__ cbH, float* __restrict__ cbF,
                         float* __restrict__ cc,
                         _Float16* __restrict__ wdnT, _Float16* __restrict__ wupT) {
  int i = blockIdx.x * blockDim.x + threadIdx.x;
  if (i < G_ * KC_) {
    float v[CD_]; float ss = 0.f;
    #pragma unroll
    for (int j = 0; j < CD_; ++j) { v[j] = cb[(size_t)i * CD_ + j]; ss += v[j] * v[j]; }
    float nrm = fmaxf(sqrtf(ss), 1e-12f);
    float s2 = 0.f;
    #pragma unroll
    for (int j = 0; j < CD_; ++j) {
      float vn = v[j] / nrm;
      cbF[(size_t)i * CD_ + j] = vn;
      cbH[(size_t)i * CD_ + j] = (_Float16)vn;
      s2 += vn * vn;
    }
    cc[i] = s2;
  }
  int iw = i - G_ * KC_;
  if (iw >= 0 && iw < G_ * CD_ * D_) {     // wdnT[g][cd][k] = Wd[g][k][cd]
    int g = iw / (CD_ * D_); int r = iw % (CD_ * D_);
    int cd = r / D_; int k = r % D_;
    wdnT[iw] = (_Float16)Wd[((size_t)g * D_ + k) * CD_ + cd];
  }
  int iu = iw - G_ * CD_ * D_;
  if (iu >= 0 && iu < G_ * D_ * CD_) {     // wupT[g][d][k] = Wu[g][k][d]
    int g = iu / (D_ * CD_); int r = iu % (D_ * CD_);
    int d = r / CD_; int k = r % CD_;
    wupT[iu] = (_Float16)Wu[((size_t)g * CD_ + k) * D_ + d];
  }
}

// ---------------- kernel 1: fused main ----------------
__global__ void __launch_bounds__(NWAVE * 32)
pvq_main(const float* __restrict__ ze, const float* __restrict__ bdown,
         const float* __restrict__ bup, const _Float16* __restrict__ cbH,
         const float* __restrict__ cbF, const float* __restrict__ cc,
         const _Float16* __restrict__ wdnT, const _Float16* __restrict__ wupT,
         float* __restrict__ out, float* __restrict__ lossPart) {
  extern __shared__ char smem[];
  _Float16* cbLds = (_Float16*)smem;               // 8192 halves (16KB)
  float*    ccLds = (float*)(smem + 16384);        // 1024 floats (4KB)
  char*     waveBase = smem + 16384 + 4096;

  const int tid = threadIdx.x;
  const int wv = tid >> 5;
  const int l = tid & 31;
  const int half = l >> 4;
  const int ln = l & 15;

  const int g  = blockIdx.x / BLKS_PER_G;
  const int rb = blockIdx.x % BLKS_PER_G;
  const int row0 = rb * ROWS_PER_BLK + wv * 16;
  const int ov = g >> 1;
  const int cbase = (g & 1) * 32;

  _Float16* inTile = (_Float16*)(waveBase + wv * WAVE_BYTES);          // 16x256 f16, [m][d]
  float*    outBuf = (float*)(waveBase + wv * WAVE_BYTES);             // 16x128 f32 (union)
  float*    zdn    = (float*)(waveBase + wv * WAVE_BYTES + 8192);      // 16x8 f32
  int*      codes  = (int*)(waveBase + wv * WAVE_BYTES + 8704);        // 16 ints

  const h8v  zero8  = {};
  const h16v zero16 = {};

  // ---- stage normalized f16 codebook + |c|^2 for this group into LDS ----
  {
    const uint4* src = (const uint4*)(cbH + (size_t)g * KC_ * CD_);    // 1024 x uint4
    uint4* dst = (uint4*)cbLds;
    for (int i = tid; i < 1024; i += NWAVE * 32) dst[i] = src[i];
    const float* cs = cc + (size_t)g * KC_;
    for (int i = tid; i < 1024; i += NWAVE * 32) ccLds[i] = cs[i];
  }
  __syncthreads();

  // ---- load 16x256 input tile into LDS in d-order (d = c'*8 + h) ----
  // zg[b,t,g,d] = z_e[b, h*W + t*OV + ov, cbase + c']
  for (int it = 0; it < 16; ++it) {
    int s = it * 4 + (l >> 3);          // (m, h-pair) segment 0..63
    int m = s >> 2, h0 = (s & 3) * 2;
    int row = row0 + m; int b = row / T_; int t = row - b * T_;
    int c4 = (l & 7) * 4;
    size_t base = ((size_t)b * HW_ + (size_t)h0 * W_ + t * OV_ + ov) * C_ + cbase + c4;
    const float4 v0 = *(const float4*)(ze + base);                    // row h0   (128B/seg)
    const float4 v1 = *(const float4*)(ze + base + (size_t)W_ * C_);  // row h0+1 (128B/seg)
    _Float16* dst = inTile + m * 256 + h0;
    h2v p;
    p[0] = (_Float16)v0.x; p[1] = (_Float16)v1.x; *(h2v*)(dst + (c4 + 0) * 8) = p;
    p[0] = (_Float16)v0.y; p[1] = (_Float16)v1.y; *(h2v*)(dst + (c4 + 1) * 8) = p;
    p[0] = (_Float16)v0.z; p[1] = (_Float16)v1.z; *(h2v*)(dst + (c4 + 2) * 8) = p;
    p[0] = (_Float16)v0.w; p[1] = (_Float16)v1.w; *(h2v*)(dst + (c4 + 3) * 8) = p;
  }
  __syncthreads();

  // ---- down projection: (16x256) @ (256x8) via 8 x WMMA f16 K=32 ----
  f8v acc = {};
  const _Float16* wdnG = wdnT + (size_t)g * CD_ * D_ + (size_t)(ln & 7) * D_;
  for (int kk = 0; kk < 8; ++kk) {
    // A: rows m=ln; elements 0..15 -> K=half*16+j (contiguous in d)
    h8v alo = *(const h8v*)(inTile + ln * 256 + kk * 32 + half * 16);
    h8v ahi = *(const h8v*)(inTile + ln * 256 + kk * 32 + half * 16 + 8);
    h16v a = SHUF16(alo, ahi);
    // B: column n=ln (cd); 16 contiguous halves, zero for padding lanes
    h16v bw = *(const h16v*)(wdnG + kk * 32 + half * 16);
    h16v bm = (ln < 8) ? bw : zero16;
    acc = __builtin_amdgcn_wmma_f32_16x16x32_f16(false, a, false, bm, (short)0, acc, false, false);
  }
  // bias + l2norm (lane = cd, VGPR = row); butterfly over cd lanes 0..7
  float bd = (ln < 8) ? bdown[g * CD_ + ln] : 0.0f;
  float zv[8];
  #pragma unroll
  for (int r = 0; r < 8; ++r) {
    float v = acc[r] + bd;
    float sq = v * v;
    sq += __shfl_xor(sq, 1, 32);
    sq += __shfl_xor(sq, 2, 32);
    sq += __shfl_xor(sq, 4, 32);
    float nrm = fmaxf(sqrtf(sq), 1e-12f);
    zv[r] = v / nrm;
  }
  if (ln < 8) {
    #pragma unroll
    for (int r = 0; r < 8; ++r) zdn[(r + half * 8) * 8 + ln] = zv[r];
  }
  __syncthreads();

  // ---- z_e_downs flush: one float4 per lane, coalesced ----
  {
    int m = l >> 1, q = (l & 1) * 4;
    int row = row0 + m; int b = row / T_; int t = row - b * T_;
    float4 zq4 = *(const float4*)(zdn + m * 8 + q);
    *(float4*)(out + OUT_ZED + (((size_t)b * G_ + g) * T_ + t) * CD_ + q) = zq4;
  }

  // ---- nearest-code search: 64 tiles of 16 codes, WMMA f16 (K padded 8->32) ----
  h16v as;
  {
    float4 z0 = *(const float4*)(zdn + ln * 8);
    float4 z1 = *(const float4*)(zdn + ln * 8 + 4);
    h16v t = zero16;
    t[0] = (_Float16)z0.x; t[1] = (_Float16)z0.y; t[2] = (_Float16)z0.z; t[3] = (_Float16)z0.w;
    t[4] = (_Float16)z1.x; t[5] = (_Float16)z1.y; t[6] = (_Float16)z1.z; t[7] = (_Float16)z1.w;
    as = half ? zero16 : t;
  }
  float bestV[8]; int bestI[8];
  #pragma unroll
  for (int r = 0; r < 8; ++r) { bestV[r] = 3.4e38f; bestI[r] = 0; }
  for (int nt = 0; nt < 64; ++nt) {
    int n = nt * 16 + ln;
    h8v cb8 = *(const h8v*)(cbLds + n * CD_);      // ds_load_b128
    h16v bs = SHUF16(cb8, zero8);
    bs = half ? zero16 : bs;
    f8v sacc = {};
    sacc = __builtin_amdgcn_wmma_f32_16x16x32_f16(false, as, false, bs, (short)0, sacc, false, false);
    float ccv = ccLds[n];
    #pragma unroll
    for (int r = 0; r < 8; ++r) {
      float d2 = ccv - 2.0f * sacc[r];     // zz is constant per row -> argmin unchanged
      if (d2 < bestV[r]) { bestV[r] = d2; bestI[r] = n; }
    }
  }
  #pragma unroll
  for (int r = 0; r < 8; ++r) {            // min across the 16 code-lanes of each half
    #pragma unroll
    for (int msk = 1; msk < 16; msk <<= 1) {
      float ovv = __shfl_xor(bestV[r], msk, 32);
      int   oii = __shfl_xor(bestI[r], msk, 32);
      if (ovv < bestV[r] || (ovv == bestV[r] && oii < bestI[r])) { bestV[r] = ovv; bestI[r] = oii; }
    }
  }
  if (ln == 0) {
    #pragma unroll
    for (int r = 0; r < 8; ++r) {
      int mrow = r + half * 8;
      codes[mrow] = bestI[r];
      int row = row0 + mrow; int b = row / T_; int t = row - b * T_;
      out[OUT_IDX + ((size_t)b * G_ + g) * T_ + t] = (float)bestI[r];      // indices
    }
  }
  __syncthreads();

  // ---- loss partial (fixed order within wave -> deterministic) ----
  float part;
  {
    int lc = l & 15;
    int code = codes[lc];
    const float4* cf4 = (const float4*)(cbF + ((size_t)g * KC_ + code) * CD_);
    float4 c0 = cf4[0], c1 = cf4[1];
    const float4* zr4 = (const float4*)(zdn + lc * 8);
    float4 a0 = zr4[0], a1 = zr4[1];
    float s = 0.f;
    float dX;
    dX = a0.x - c0.x; s += dX * dX;  dX = a0.y - c0.y; s += dX * dX;
    dX = a0.z - c0.z; s += dX * dX;  dX = a0.w - c0.w; s += dX * dX;
    dX = a1.x - c1.x; s += dX * dX;  dX = a1.y - c1.y; s += dX * dX;
    dX = a1.z - c1.z; s += dX * dX;  dX = a1.w - c1.w; s += dX * dX;
    part = (l < 16) ? s : 0.f;
  }
  part += __shfl_xor(part, 1, 32);
  part += __shfl_xor(part, 2, 32);
  part += __shfl_xor(part, 4, 32);
  part += __shfl_xor(part, 8, 32);
  part += __shfl_xor(part, 16, 32);
  if (l == 0) lossPart[blockIdx.x * NWAVE + wv] = part;

  // ---- up projection: (16x8) @ (8x256) via WMMA, permuted N-tiles for 128B stores ----
  h16v au;
  {
    int code = codes[ln];
    h8v c8 = *(const h8v*)(cbLds + code * CD_);    // straight-through value = cb[code]
    h16v t = SHUF16(c8, zero8);
    au = half ? zero16 : t;
  }
  const _Float16* wupG = wupT + (size_t)g * D_ * CD_;
  for (int hb = 0; hb < 2; ++hb) {
    for (int ut = 0; ut < 8; ++ut) {
      int cpr = ut * 4 + (ln >> 2);
      int hmod = ln & 3;
      int d = cpr * 8 + hb * 4 + hmod;             // column of W_up for this lane
      h8v w8 = *(const h8v*)(wupG + d * CD_);      // 16B contiguous
      h16v bu = SHUF16(w8, zero8);
      bu = half ? zero16 : bu;
      f8v u = {};
      u = __builtin_amdgcn_wmma_f32_16x16x32_f16(false, au, false, bu, (short)0, u, false, false);
      float bv = bup[g * D_ + d];
      #pragma unroll
      for (int r = 0; r < 8; ++r) {
        int mrow = r + half * 8;
        outBuf[mrow * 128 + ut * 16 + ln] = u[r] + bv;   // [m][c'*4 + hmod]
      }
    }
    __syncthreads();
    // flush half-tile: full 128B coalesced stores of z_q
    for (int it = 0; it < 16; ++it) {
      int s = it * 4 + (l >> 3);          // (m, hmod) segment 0..63
      int m = s >> 2, hmod = s & 3;
      int h = hb * 4 + hmod;
      int c4 = (l & 7) * 4;
      float4 v;
      v.x = outBuf[m * 128 + (c4 + 0) * 4 + hmod];
      v.y = outBuf[m * 128 + (c4 + 1) * 4 + hmod];
      v.z = outBuf[m * 128 + (c4 + 2) * 4 + hmod];
      v.w = outBuf[m * 128 + (c4 + 3) * 4 + hmod];
      int row = row0 + m; int b = row / T_; int t = row - b * T_;
      *(float4*)(out + ((size_t)b * HW_ + (size_t)h * W_ + t * OV_ + ov) * C_ + cbase + c4) = v;
    }
    __syncthreads();
  }
}

// ---------------- kernel 2: deterministic loss reduction ----------------
__global__ void pvq_loss(const float* __restrict__ part, float* __restrict__ out) {
  __shared__ float red[256];
  float s = 0.0f;
  for (int i = threadIdx.x; i < NPART; i += 256) s += part[i];
  red[threadIdx.x] = s;
  __syncthreads();
  for (int st = 128; st > 0; st >>= 1) {
    if (threadIdx.x < st) red[threadIdx.x] += red[threadIdx.x + st];
    __syncthreads();
  }
  if (threadIdx.x == 0) {
    float mean = red[0] * (1.0f / (float)((size_t)B_ * T_ * G_ * CD_));   // /1024000
    out[OUT_CL] = mean;   // commitment_loss
    out[OUT_CB] = mean;   // codebook_loss (same forward value)
  }
}

// ---------------- launcher ----------------
extern "C" void kernel_launch(void* const* d_in, const int* in_sizes, int n_in,
                              void* d_out, int out_size, void* d_ws, size_t ws_size,
                              hipStream_t stream) {
  (void)in_sizes; (void)n_in; (void)out_size; (void)ws_size;
  const float* ze = (const float*)d_in[0];
  const float* Wd = (const float*)d_in[1];
  const float* bd = (const float*)d_in[2];
  const float* Wu = (const float*)d_in[3];
  const float* bu = (const float*)d_in[4];
  const float* cb = (const float*)d_in[5];
  char* ws = (char*)d_ws;
  _Float16* cbH  = (_Float16*)(ws + WS_CBH);
  float*    cbF  = (float*)(ws + WS_CBF);
  float*    ccv  = (float*)(ws + WS_CC);
  _Float16* wdnT = (_Float16*)(ws + WS_WDN);
  _Float16* wupT = (_Float16*)(ws + WS_WUP);
  float*    lossPart = (float*)(ws + WS_LOSS);
  float* out = (float*)d_out;

  pvq_prep<<<160, 256, 0, stream>>>(Wd, Wu, cb, cbH, cbF, ccv, wdnT, wupT);
  pvq_main<<<NBLK, NWAVE * 32, SMEM_BYTES, stream>>>(ze, bd, bu, cbH, cbF, ccv,
                                                     wdnT, wupT, out, lossPart);
  pvq_loss<<<1, 256, 0, stream>>>(lossPart, out);
}
